// StudentMoELayer_51453708206111
// MI455X (gfx1250) — compile-verified, matrix operand order checked by reading, not run on hardware
//
#include <hip/hip_runtime.h>
#include <hip/hip_bf16.h>
#include <math.h>

// ---------------- problem constants ----------------
#define NE     8          // experts
#define TOPK   2
#define DMODEL 2048
#define FE     1024       // per-expert intermediate
#define TTOK   4096       // B*S tokens
#define NPAIR  (TTOK * TOPK)

typedef __bf16 bf16_t;
typedef __attribute__((ext_vector_type(16))) __bf16 v16bf;
typedef __attribute__((ext_vector_type(8)))  float  v8f;

// Load a 16-element bf16 WMMA fragment (32 bytes) as two b128 loads.
__device__ __forceinline__ v16bf load_frag(const bf16_t* p) {
    union { v16bf v; uint4 q[2]; } f;
    const uint4* q = (const uint4*)p;
    f.q[0] = q[0];
    f.q[1] = q[1];
    return f.v;
}

__device__ __forceinline__ unsigned int pack_bf16(float a, float b) {
    union { __bf16 h[2]; unsigned int u; } p;
    p.h[0] = (__bf16)a;
    p.h[1] = (__bf16)b;
    return p.u;
}

// ---------------------------------------------------------------------------
// Kernel 1: RMSNorm(router input) + router logits + top-2 selection.
// One wave (32 lanes) per token.
// ---------------------------------------------------------------------------
__global__ __launch_bounds__(256) void moe_router(
    const float* __restrict__ x,      // [T, D]
    const float* __restrict__ rw,     // [E, D]
    const float* __restrict__ nw,     // [D]
    int*   __restrict__ counts,       // [E]
    int*   __restrict__ tope,         // [T, 2]
    float* __restrict__ topw)         // [T, 2]
{
    const int wave = threadIdx.x >> 5;
    const int lane = threadIdx.x & 31;
    const int t = blockIdx.x * 8 + wave;
    if (t >= TTOK) return;

    const float* xr = x + (size_t)t * DMODEL;
    float ss = 0.f;
    float acc[NE];
#pragma unroll
    for (int e = 0; e < NE; ++e) acc[e] = 0.f;

    for (int d = lane; d < DMODEL; d += 32) {
        float xv = xr[d];
        float xn = xv * nw[d];
        ss += xv * xv;
#pragma unroll
        for (int e = 0; e < NE; ++e) acc[e] += xn * rw[e * DMODEL + d];
    }
#pragma unroll
    for (int off = 16; off > 0; off >>= 1) {
        ss += __shfl_xor(ss, off, 32);
#pragma unroll
        for (int e = 0; e < NE; ++e) acc[e] += __shfl_xor(acc[e], off, 32);
    }

    if (lane == 0) {
        float r = rsqrtf(ss * (1.f / (float)DMODEL) + 1e-6f);
        float lg[NE];
#pragma unroll
        for (int e = 0; e < NE; ++e) lg[e] = r * acc[e];
        int i0 = 0; float l0 = lg[0];
#pragma unroll
        for (int e = 1; e < NE; ++e) if (lg[e] > l0) { l0 = lg[e]; i0 = e; }
        int i1 = (i0 == 0) ? 1 : 0; float l1 = lg[i1];
#pragma unroll
        for (int e = 0; e < NE; ++e)
            if (e != i0 && e != i1 && lg[e] > l1) { l1 = lg[e]; i1 = e; }
        float w0 = 1.f / (1.f + expf(l1 - l0));
        float w1 = 1.f - w0;
        tope[2 * t + 0] = i0;  topw[2 * t + 0] = w0 * 4.0f;  // SCALE=4 folded
        tope[2 * t + 1] = i1;  topw[2 * t + 1] = w1 * 4.0f;
        atomicAdd(&counts[i0], 1);
        atomicAdd(&counts[i1], 1);
    }
}

// ---------------------------------------------------------------------------
// Kernel 2: exclusive scan over 8 expert counts -> segment offsets + cursors
// ---------------------------------------------------------------------------
__global__ void moe_scan(const int* __restrict__ counts,
                         int* __restrict__ offsets,
                         int* __restrict__ cursor)
{
    if (threadIdx.x == 0) {
        int run = 0;
#pragma unroll
        for (int e = 0; e < NE; ++e) {
            offsets[e] = run;
            cursor[e]  = run;
            run += counts[e];
        }
    }
}

// ---------------------------------------------------------------------------
// Kernel 3: compact (token, weight) pairs into per-expert segments
// ---------------------------------------------------------------------------
__global__ __launch_bounds__(256) void moe_fill(
    const int*   __restrict__ tope,
    const float* __restrict__ topw,
    int*   __restrict__ cursor,
    int*   __restrict__ list,
    float* __restrict__ wgt)
{
    int t = blockIdx.x * blockDim.x + threadIdx.x;
    if (t >= TTOK) return;
#pragma unroll
    for (int k = 0; k < TOPK; ++k) {
        int e = tope[2 * t + k];
        int pos = atomicAdd(&cursor[e], 1);
        list[pos] = t;
        wgt[pos]  = topw[2 * t + k];
    }
}

// ---------------------------------------------------------------------------
// Kernel 4a: swizzle hidden states f32 -> bf16 in WMMA A-fragment order.
// dst[t][kb][32] ; entry (half*16 + j*2 + s) = x[t][kb*32 + koff(j,half)+s]
// One thread per (t, kb); source 32 floats are contiguous.
// ---------------------------------------------------------------------------
__global__ __launch_bounds__(256) void swz_x(
    const float* __restrict__ x, bf16_t* __restrict__ dst)
{
    int tid = blockIdx.x * 256 + threadIdx.x;      // T*64 threads
    int t  = tid >> 6;
    int kb = tid & 63;
    const float* s = x + (size_t)t * DMODEL + kb * 32;
    union { uint4 q[4]; unsigned int u[16]; } o;
#pragma unroll
    for (int m = 0; m < 16; ++m) {
        int half = m >> 3, j = m & 7;
        int k0 = ((j >> 2) << 4) + (half << 3) + ((j & 3) << 1);
        o.u[m] = pack_bf16(s[k0], s[k0 + 1]);
    }
    uint4* d = (uint4*)(dst + (size_t)tid * 32);
    d[0] = o.q[0]; d[1] = o.q[1]; d[2] = o.q[2]; d[3] = o.q[3];
}

// ---------------------------------------------------------------------------
// Kernel 4b: swizzle a weight matrix [EKB*32, N] f32 -> bf16 fragment order.
// dst[ekb][n][32]. One thread per (ekb, n); loads are coalesced across n.
// ---------------------------------------------------------------------------
__global__ __launch_bounds__(256) void swz_weights(
    const float* __restrict__ src, bf16_t* __restrict__ dst,
    int nShift /* log2(N) */, int total)
{
    int tid = blockIdx.x * 256 + threadIdx.x;
    if (tid >= total) return;
    const int N = 1 << nShift;
    int n   = tid & (N - 1);
    int ekb = tid >> nShift;
    const float* s = src + (size_t)ekb * 32 * N + n;
    union { uint4 q[4]; unsigned int u[16]; } o;
#pragma unroll
    for (int m = 0; m < 16; ++m) {
        int half = m >> 3, j = m & 7;
        int k0 = ((j >> 2) << 4) + (half << 3) + ((j & 3) << 1);
        o.u[m] = pack_bf16(s[(size_t)k0 * N], s[(size_t)(k0 + 1) * N]);
    }
    uint4* d = (uint4*)(dst + (size_t)tid * 32);
    d[0] = o.q[0]; d[1] = o.q[1]; d[2] = o.q[2]; d[3] = o.q[3];
}

// ---------------------------------------------------------------------------
// Kernel 5: grouped gate/up GEMM + SiLU*up. bf16 WMMA, f32 accumulate.
// Workgroup: 64 rows x 128 F cols, 8 waves as 2(M) x 4(N).
// Wave: 32 rows x 32 cols -> 8 accumulators; A frags reused across B frags.
// Inner loop per wave: 4 A b128 + 8 B b128 loads, 8 WMMAs.
// ---------------------------------------------------------------------------
__global__ __launch_bounds__(256) void moe_gateup(
    const bf16_t* __restrict__ xs,     // swizzled x   [T*D]
    const bf16_t* __restrict__ gws,    // swizzled gate [E*D*FE]
    const bf16_t* __restrict__ uws,    // swizzled up   [E*D*FE]
    const int*    __restrict__ counts,
    const int*    __restrict__ offsets,
    const int*    __restrict__ list,
    bf16_t*       __restrict__ act)    // swizzled activations [2T*FE]
{
    const int e  = blockIdx.z;
    const int mb = blockIdx.y;
    const int fb = blockIdx.x;
    const int ne = counts[e];
    if (mb * 64 >= ne) return;
    const int base = offsets[e];

    const int wave = threadIdx.x >> 5, lane = threadIdx.x & 31;
    const int mw = wave & 1, nwv = wave >> 1;     // 2 x 4 wave grid
    const int m0 = mb * 64 + mw * 32;
    const int f0 = fb * 128 + nwv * 32;
    const int half = lane >> 4, l15 = lane & 15;

    int r0 = m0 + l15;       int c0 = (r0 < ne) ? r0 : ne - 1;
    int r1 = m0 + 16 + l15;  int c1 = (r1 < ne) ? r1 : ne - 1;
    const bf16_t* a0p = xs + (size_t)list[base + c0] * DMODEL + half * 16;
    const bf16_t* a1p = xs + (size_t)list[base + c1] * DMODEL + half * 16;

    // weight fragment base: ((e*64 + kb)*1024 + n)*32 + half*16
    size_t wb = ((size_t)e * 64 * FE + (f0 + l15)) * 32 + half * 16;
    const bf16_t* bg = gws + wb;
    const bf16_t* bu = uws + wb;

    v8f cg[2][2] = {{{}, {}}, {{}, {}}};   // [Afrag][Ntile]
    v8f cu[2][2] = {{{}, {}}, {{}, {}}};

    for (int kb = 0; kb < 64; ++kb) {
        v16bf a0 = load_frag(a0p + kb * 32);
        v16bf a1 = load_frag(a1p + kb * 32);
        v16bf b;
        b = load_frag(bg);
        cg[0][0] = __builtin_amdgcn_wmma_f32_16x16x32_bf16(false, a0, false, b, (short)0, cg[0][0], false, false);
        cg[1][0] = __builtin_amdgcn_wmma_f32_16x16x32_bf16(false, a1, false, b, (short)0, cg[1][0], false, false);
        b = load_frag(bg + 512);
        cg[0][1] = __builtin_amdgcn_wmma_f32_16x16x32_bf16(false, a0, false, b, (short)0, cg[0][1], false, false);
        cg[1][1] = __builtin_amdgcn_wmma_f32_16x16x32_bf16(false, a1, false, b, (short)0, cg[1][1], false, false);
        b = load_frag(bu);
        cu[0][0] = __builtin_amdgcn_wmma_f32_16x16x32_bf16(false, a0, false, b, (short)0, cu[0][0], false, false);
        cu[1][0] = __builtin_amdgcn_wmma_f32_16x16x32_bf16(false, a1, false, b, (short)0, cu[1][0], false, false);
        b = load_frag(bu + 512);
        cu[0][1] = __builtin_amdgcn_wmma_f32_16x16x32_bf16(false, a0, false, b, (short)0, cu[0][1], false, false);
        cu[1][1] = __builtin_amdgcn_wmma_f32_16x16x32_bf16(false, a1, false, b, (short)0, cu[1][1], false, false);
        bg += (size_t)FE * 32;
        bu += (size_t)FE * 32;
    }

    // epilogue: h = silu(g)*u, stored directly in swizzled A-fragment order
    // within-32 entry index for col (nt*16 + l15): nt*8 + l15 + (l15 & 8)
    const int idx0 = l15 + (l15 & 8);
#pragma unroll
    for (int fi = 0; fi < 2; ++fi) {
#pragma unroll
        for (int r = 0; r < 8; ++r) {
            int row = m0 + fi * 16 + half * 8 + r;
            if (row < ne) {
                size_t ob = (size_t)(base + row) * FE + f0;
                float g = cg[fi][0][r], u = cu[fi][0][r];
                act[ob + idx0] = (__bf16)((g / (1.f + expf(-g))) * u);
                g = cg[fi][1][r]; u = cu[fi][1][r];
                act[ob + 8 + idx0] = (__bf16)((g / (1.f + expf(-g))) * u);
            }
        }
    }
}

// ---------------------------------------------------------------------------
// Kernel 6: grouped down-proj GEMM, combine-weight scale, atomic scatter-add.
// Same tiling: workgroup 64 rows x 128 D cols, wave 32x32.
// ---------------------------------------------------------------------------
__global__ __launch_bounds__(256) void moe_down(
    const bf16_t* __restrict__ act,    // swizzled [2T*FE]
    const bf16_t* __restrict__ dws,    // swizzled down_w [E*FE*D]
    const int*    __restrict__ counts,
    const int*    __restrict__ offsets,
    const int*    __restrict__ list,
    const float*  __restrict__ wgt,
    float*        __restrict__ out)    // [T, D] (pre-zeroed)
{
    const int e  = blockIdx.z;
    const int mb = blockIdx.y;
    const int db = blockIdx.x;
    const int ne = counts[e];
    if (mb * 64 >= ne) return;
    const int base = offsets[e];

    const int wave = threadIdx.x >> 5, lane = threadIdx.x & 31;
    const int mw = wave & 1, nwv = wave >> 1;
    const int m0 = mb * 64 + mw * 32;
    const int d0 = db * 128 + nwv * 32;
    const int half = lane >> 4, l15 = lane & 15;

    int r0 = m0 + l15;       int c0 = (r0 < ne) ? r0 : ne - 1;
    int r1 = m0 + 16 + l15;  int c1 = (r1 < ne) ? r1 : ne - 1;
    const bf16_t* a0p = act + (size_t)(base + c0) * FE + half * 16;
    const bf16_t* a1p = act + (size_t)(base + c1) * FE + half * 16;

    // weight fragment base: ((e*32 + kb)*2048 + n)*32 + half*16
    const bf16_t* bw = dws + ((size_t)e * 32 * DMODEL + (d0 + l15)) * 32 + half * 16;

    v8f c[2][2] = {{{}, {}}, {{}, {}}};

    for (int kb = 0; kb < 32; ++kb) {
        v16bf a0 = load_frag(a0p + kb * 32);
        v16bf a1 = load_frag(a1p + kb * 32);
        v16bf b;
        b = load_frag(bw);
        c[0][0] = __builtin_amdgcn_wmma_f32_16x16x32_bf16(false, a0, false, b, (short)0, c[0][0], false, false);
        c[1][0] = __builtin_amdgcn_wmma_f32_16x16x32_bf16(false, a1, false, b, (short)0, c[1][0], false, false);
        b = load_frag(bw + 512);
        c[0][1] = __builtin_amdgcn_wmma_f32_16x16x32_bf16(false, a0, false, b, (short)0, c[0][1], false, false);
        c[1][1] = __builtin_amdgcn_wmma_f32_16x16x32_bf16(false, a1, false, b, (short)0, c[1][1], false, false);
        bw += (size_t)DMODEL * 32;
    }

#pragma unroll
    for (int fi = 0; fi < 2; ++fi) {
#pragma unroll
        for (int r = 0; r < 8; ++r) {
            int row = m0 + fi * 16 + half * 8 + r;
            if (row < ne) {
                int   t = list[base + row];
                float w = wgt[base + row];
                float* op = out + (size_t)t * DMODEL + d0;
                atomicAdd(op + l15,      c[fi][0][r] * w);
                atomicAdd(op + 16 + l15, c[fi][1][r] * w);
            }
        }
    }
}

// ---------------------------------------------------------------------------
// Host-side launch
// ---------------------------------------------------------------------------
extern "C" void kernel_launch(void* const* d_in, const int* in_sizes, int n_in,
                              void* d_out, int out_size, void* d_ws, size_t ws_size,
                              hipStream_t stream) {
    (void)in_sizes; (void)n_in; (void)out_size; (void)ws_size;

    const float* x   = (const float*)d_in[0];   // [2,2048,2048]
    const float* rw  = (const float*)d_in[1];   // [8,2048]
    const float* nw  = (const float*)d_in[2];   // [2048]
    const float* gw  = (const float*)d_in[3];   // [8,2048,1024]
    const float* uw  = (const float*)d_in[4];   // [8,2048,1024]
    const float* dwn = (const float*)d_in[5];   // [8,1024,2048]
    float* out = (float*)d_out;

    // ---- workspace carve-up (~135 MB) ----
    char* ws = (char*)d_ws;
    size_t off = 0;
    auto carve = [&](size_t bytes) -> char* {
        char* p = ws + off;
        off = (off + bytes + 255) & ~(size_t)255;
        return p;
    };
    int*    counts  = (int*)carve(3 * NE * sizeof(int));     // counts|cursor|offsets
    int*    cursor  = counts + NE;
    int*    offsets = cursor + NE;
    int*    tope  = (int*)  carve(NPAIR * sizeof(int));
    float*  topw  = (float*)carve(NPAIR * sizeof(float));
    int*    list  = (int*)  carve(NPAIR * sizeof(int));
    float*  wgt   = (float*)carve(NPAIR * sizeof(float));
    bf16_t* xs    = (bf16_t*)carve((size_t)TTOK * DMODEL * 2);        // 16.8 MB
    bf16_t* act   = (bf16_t*)carve((size_t)NPAIR * FE * 2);           // 16.8 MB
    bf16_t* gws   = (bf16_t*)carve((size_t)NE * DMODEL * FE * 2);     // 33.6 MB
    bf16_t* uws   = (bf16_t*)carve((size_t)NE * DMODEL * FE * 2);     // 33.6 MB
    bf16_t* dws   = (bf16_t*)carve((size_t)NE * FE * DMODEL * 2);     // 33.6 MB

    hipMemsetAsync(counts, 0, NE * sizeof(int), stream);
    hipMemsetAsync(out, 0, (size_t)TTOK * DMODEL * sizeof(float), stream);

    // one-pass f32 -> bf16 fragment-order swizzles (HBM streaming, ~15us)
    swz_x<<<(TTOK * 64) / 256, 256, 0, stream>>>(x, xs);
    {
        int totGU = NE * (DMODEL / 32) * FE;     // 524288
        int totD  = NE * (FE / 32) * DMODEL;     // 524288
        swz_weights<<<(totGU + 255) / 256, 256, 0, stream>>>(gw,  gws, 10, totGU);
        swz_weights<<<(totGU + 255) / 256, 256, 0, stream>>>(uw,  uws, 10, totGU);
        swz_weights<<<(totD  + 255) / 256, 256, 0, stream>>>(dwn, dws, 11, totD);
    }

    moe_router<<<TTOK / 8, 256, 0, stream>>>(x, rw, nw, counts, tope, topw);
    moe_scan<<<1, 32, 0, stream>>>(counts, offsets, cursor);
    moe_fill<<<TTOK / 256, 256, 0, stream>>>(tope, topw, cursor, list, wgt);

    dim3 g1(FE / 128, 64, NE);
    moe_gateup<<<g1, 256, 0, stream>>>(xs, gws, uws, counts, offsets, list, act);

    dim3 g2(DMODEL / 128, 64, NE);
    moe_down<<<g2, 256, 0, stream>>>(act, dws, counts, offsets, list, wgt, out);
}
